// WindowMSA_78898549228101
// MI455X (gfx1250) — matrix operand order repaired and checked by reading, not compiled, not run on hardware
//
#include <hip/hip_runtime.h>
#include <math.h>
#include <stdint.h>

// ---- problem constants ------------------------------------------------------
#define B_    4096
#define N_    49
#define C_    128
#define H_    4
#define VC_   32
#define NW_   64
#define R_    (B_ * N_)      // 200704 rows of the qkv GEMM (divisible by 16)
#define C3_   (3 * C_)       // 384
#define SCALE_ 0.17677669529663689f   // 1/sqrt(32)
#define VST_  72             // padded row stride (halves) for transposed V in LDS

typedef __attribute__((ext_vector_type(16))) _Float16 v16h;
typedef __attribute__((ext_vector_type(8)))  _Float16 v8h;
typedef __attribute__((ext_vector_type(8)))  float    v8f;

// ============================================================================
// Kernel 0: one-shot convert qkv_w (384x128 f32) -> f16, row-major.
//   98 KB, stays L2-resident for the whole GEMM.
// ============================================================================
__global__ __launch_bounds__(256) void wcvt_kernel(
    const float* __restrict__ w, _Float16* __restrict__ wh)
{
    int i = blockIdx.x * 256 + threadIdx.x;   // 49152 elements, 192 blocks
    wh[i] = (_Float16)w[i];
}

// ============================================================================
// Kernel 1: QKV projection  qkv = x @ W^T + b   (f32 in, f16 out, f32 acc)
//   grid.x = R_/16 row tiles, 256 threads = 8 waves, each wave: 3 col tiles.
// ============================================================================
__global__ __launch_bounds__(256) void qkv_gemm_kernel(
    const float* __restrict__ x, const _Float16* __restrict__ wh,
    const float* __restrict__ bias, _Float16* __restrict__ qkvh)
{
    __shared__ __align__(16) _Float16 xs[16 * 128];   // 16-row strip of x as f16

    const int tid = threadIdx.x;
    const int rowTile = blockIdx.x;

    for (int i = tid; i < 16 * 128; i += 256) {
        int r = i >> 7, k = i & 127;
        xs[i] = (_Float16)x[(long)(rowTile * 16 + r) * 128 + k];
    }
    __syncthreads();

    const int wave = tid >> 5, lane = tid & 31;
    const int ln = lane & 15, hi = lane >> 4;

    // A fragments (16x32 f16), contiguous K layout:
    //   lane ln (+hi K-group): K=16*hi+j
    v16h a[4];
#pragma unroll
    for (int kt = 0; kt < 4; ++kt) {
        const _Float16* row = &xs[ln * 128 + kt * 32];
        v8h lo  = *(const v8h*)(row + 16 * hi);       // 16B aligned
        v8h hi8 = *(const v8h*)(row + 16 * hi + 8);   // 16B aligned
        a[kt] = __builtin_shufflevector(lo, hi8,
                    0,1,2,3,4,5,6,7,8,9,10,11,12,13,14,15);
    }

#pragma unroll
    for (int i = 0; i < 3; ++i) {
        const int ct = wave * 3 + i;           // col tile 0..23
        const int c  = ct * 16 + ln;           // this lane's output column
        v8f acc = {};
#pragma unroll
        for (int kt = 0; kt < 4; ++kt) {
            // B[k][c] = wh[c*128 + k]; lane holds col c, K = kt*32+16*hi+j
            v16h bfr = *(const v16h*)&wh[(long)c * 128 + kt * 32 + 16 * hi];
            acc = __builtin_amdgcn_wmma_f32_16x16x32_f16(
                false, a[kt], false, bfr, (short)0, acc, false, false);
        }
        const float bv = bias[c];
        const float sc = (c < 128) ? SCALE_ : 1.0f;   // pre-scale q by 1/sqrt(d)
#pragma unroll
        for (int r = 0; r < 8; ++r) {                 // C/D: row = r + 8*hi
            long row_g = (long)rowTile * 16 + r + 8 * hi;
            qkvh[row_g * C3_ + c] = (_Float16)((acc[r] + bv) * sc);
        }
    }
}

// ============================================================================
// Kernel 2: windowed attention + fused LePE, one block (4 waves) per (b, h).
//   Q/K staged into LDS with GLOBAL_LOAD_ASYNC_TO_LDS_B128 (ASYNCcnt path,
//   no VGPR round-trip); V goes through VGPRs because it is transposed into
//   vsT[d][k] (stride 72) so the WMMA B-fragment and LePE taps are contiguous.
// ============================================================================
__global__ __launch_bounds__(128) void attn_kernel(
    const _Float16* __restrict__ qkvh, const float* __restrict__ mask,
    const float* __restrict__ lw, const float* __restrict__ lb,
    float* __restrict__ out)
{
    __shared__ __align__(16) _Float16 qs[64 * 32];
    __shared__ __align__(16) _Float16 ks[64 * 32];
    __shared__ __align__(16) _Float16 vsT[VC_ * VST_]; // vsT[d*72+n] = v[n][d]
    __shared__ __align__(16) _Float16 ps[4][16 * 64];  // per-wave P strip
    __shared__ float lwsh[VC_ * 9];
    __shared__ float lbsh[VC_];

    const int tid = threadIdx.x;
    const int bh = blockIdx.x;
    const int b = bh >> 2, h = bh & 3;
    const long rowbase = (long)b * N_;
    const long maskoff = (long)(b & (NW_ - 1)) * (N_ * N_);

    // --- async copy of Q and K valid rows (49 x 32 halves = 196 16B chunks) ---
    // LDS dest offset = low 32 bits of the flat shared pointer (ISA 10.2).
    for (int i = tid; i < (N_ * 32) / 8; i += 128) {
        int n = i >> 2, ch = i & 3;                       // row, 16B chunk
        const _Float16* gq = qkvh + (rowbase + n) * C3_ + h * 32 + ch * 8;
        const _Float16* gk = gq + 128;
        unsigned lq = (unsigned)(uintptr_t)&qs[n * 32 + ch * 8];
        unsigned lk = (unsigned)(uintptr_t)&ks[n * 32 + ch * 8];
        asm volatile("global_load_async_to_lds_b128 %0, %1, off"
                     :: "v"(lq), "v"(gq) : "memory");
        asm volatile("global_load_async_to_lds_b128 %0, %1, off"
                     :: "v"(lk), "v"(gk) : "memory");
    }

    // zero-fill ONLY the padding rows (disjoint from the async-written region)
    for (int i = N_ * 32 + tid; i < 64 * 32; i += 128) {
        qs[i] = (_Float16)0.f; ks[i] = (_Float16)0.f;
    }
    for (int i = tid; i < VC_ * VST_; i += 128) vsT[i] = (_Float16)0.f;
    for (int i = tid; i < VC_ * 9; i += 128)
        lwsh[i] = lw[(h * VC_ + i / 9) * 9 + i % 9];
    if (tid < VC_) lbsh[tid] = lb[h * VC_ + tid];
    __syncthreads();

    // V through VGPRs (transposed store); overlaps with outstanding async Q/K
    for (int i = tid; i < N_ * 32; i += 128) {
        int n = i >> 5, d = i & 31;
        vsT[d * VST_ + n] = qkvh[(rowbase + n) * C3_ + h * 32 + d + 256];
    }
    asm volatile("s_wait_asynccnt 0x0" ::: "memory");   // Q/K resident in LDS
    __syncthreads();

    const int wave = tid >> 5, lane = tid & 31;
    const int ln = lane & 15, hi = lane >> 4;

    // A = Q tile (rows wave*16 .. +15, K = 0..31)
    v16h aq;
    {
        const _Float16* row = &qs[(wave * 16 + ln) * 32];
        v8h lo  = *(const v8h*)(row + 16 * hi);
        v8h hi8 = *(const v8h*)(row + 16 * hi + 8);
        aq = __builtin_shufflevector(lo, hi8,
                 0,1,2,3,4,5,6,7,8,9,10,11,12,13,14,15);
    }

    // S = Q K^T : B[k][n] = ks[nt*16+n][k]; key rows contiguous in K
    v8f acc[4];
#pragma unroll
    for (int nt = 0; nt < 4; ++nt) {
        v16h bk = *(const v16h*)&ks[(nt * 16 + ln) * 32 + 16 * hi];
        v8f z = {};
        acc[nt] = __builtin_amdgcn_wmma_f32_16x16x32_f16(
            false, aq, false, bk, (short)0, z, false, false);
    }

    // mask + padding handling
#pragma unroll
    for (int nt = 0; nt < 4; ++nt) {
        int ng = nt * 16 + ln;
#pragma unroll
        for (int r = 0; r < 8; ++r) {
            int mg = wave * 16 + r + 8 * hi;
            float s = acc[nt][r];
            if (mg >= N_)      s = 0.0f;       // dead row (discarded at store)
            else if (ng >= N_) s = -1e30f;     // padded key column -> exp = 0
            else               s += mask[maskoff + mg * N_ + ng];
            acc[nt][r] = s;
        }
    }

    // row softmax: a C-matrix row lives in one 16-lane half -> xor-reduce 1..8
#pragma unroll
    for (int r = 0; r < 8; ++r) {
        float mx = fmaxf(fmaxf(acc[0][r], acc[1][r]), fmaxf(acc[2][r], acc[3][r]));
#pragma unroll
        for (int off = 1; off <= 8; off <<= 1) mx = fmaxf(mx, __shfl_xor(mx, off, 32));
        float e[4], sum = 0.f;
#pragma unroll
        for (int nt = 0; nt < 4; ++nt) { e[nt] = __expf(acc[nt][r] - mx); sum += e[nt]; }
#pragma unroll
        for (int off = 1; off <= 8; off <<= 1) sum += __shfl_xor(sum, off, 32);
        float inv = 1.0f / sum;
#pragma unroll
        for (int nt = 0; nt < 4; ++nt) acc[nt][r] = e[nt] * inv;
    }

    // write P strip to LDS to re-layout C/D -> A fragments
    _Float16* pw = &ps[wave][0];
#pragma unroll
    for (int nt = 0; nt < 4; ++nt)
#pragma unroll
        for (int r = 0; r < 8; ++r)
            pw[(r + 8 * hi) * 64 + nt * 16 + ln] = (_Float16)acc[nt][r];
    __syncthreads();

    // O = P V : A = P (16x64, two K-steps), B = V via vsT (contiguous rows)
    v16h ap[2];
#pragma unroll
    for (int ksz = 0; ksz < 2; ++ksz) {
        const _Float16* row = &pw[ln * 64 + ksz * 32];
        v8h lo  = *(const v8h*)(row + 16 * hi);
        v8h hi8 = *(const v8h*)(row + 16 * hi + 8);
        ap[ksz] = __builtin_shufflevector(lo, hi8,
                      0,1,2,3,4,5,6,7,8,9,10,11,12,13,14,15);
    }
#pragma unroll
    for (int nt = 0; nt < 2; ++nt) {
        const int dl = nt * 16 + ln;           // channel within head (0..31)
        v8f o = {};
#pragma unroll
        for (int ksz = 0; ksz < 2; ++ksz) {
            // B[k][d] = vsT[d*72 + k], K = ksz*32 + 16*hi + j (contiguous)
            v16h bv = *(const v16h*)&vsT[dl * VST_ + ksz * 32 + 16 * hi];
            o = __builtin_amdgcn_wmma_f32_16x16x32_f16(
                false, ap[ksz], false, bv, (short)0, o, false, false);
        }
        // fused LePE: depthwise 3x3 over this head's V slice (in LDS)
#pragma unroll
        for (int r = 0; r < 8; ++r) {
            int mg = wave * 16 + r + 8 * hi;
            if (mg < N_) {
                int y = mg / 7, x0 = mg % 7;
                float lep = lbsh[dl];
#pragma unroll
                for (int ky = 0; ky < 3; ++ky)
#pragma unroll
                    for (int kx = 0; kx < 3; ++kx) {
                        int yy = y + ky - 1, xx = x0 + kx - 1;
                        if (yy >= 0 && yy < 7 && xx >= 0 && xx < 7)
                            lep += (float)vsT[dl * VST_ + yy * 7 + xx]
                                   * lwsh[dl * 9 + ky * 3 + kx];
                    }
                out[(rowbase + mg) * C_ + h * 32 + dl] = o[r] + lep;
            }
        }
    }
}

// ============================================================================
extern "C" void kernel_launch(void* const* d_in, const int* in_sizes, int n_in,
                              void* d_out, int out_size, void* d_ws, size_t ws_size,
                              hipStream_t stream) {
    const float* x      = (const float*)d_in[0];
    const float* mask   = (const float*)d_in[1];
    const float* qkv_w  = (const float*)d_in[2];
    const float* qkv_b  = (const float*)d_in[3];
    const float* lepe_w = (const float*)d_in[4];
    const float* lepe_b = (const float*)d_in[5];
    float* out = (float*)d_out;

    _Float16* qkvh = (_Float16*)d_ws;                 // (B*N, 384) f16, ~147 MB
    _Float16* wh   = qkvh + (size_t)R_ * C3_;         // (384, 128) f16, 98 KB

    wcvt_kernel<<<(C3_ * C_) / 256, 256, 0, stream>>>(qkv_w, wh);
    qkv_gemm_kernel<<<R_ / 16, 256, 0, stream>>>(x, wh, qkv_b, qkvh);
    attn_kernel<<<B_ * H_, 128, 0, stream>>>(qkvh, mask, lepe_w, lepe_b, out);
}
